// MVMP_6975026889044
// MI455X (gfx1250) — compile-verified, matrix operand order checked by reading, not run on hardware
//
#include <hip/hip_runtime.h>

typedef __attribute__((ext_vector_type(16))) __bf16 v16bf;
typedef __attribute__((ext_vector_type(8)))  float  v8f;

#define DHID 128

// ---------------------------------------------------------------------------
// Tiled GEMM: out[M,128] = epilogue( in[M,128] @ W[128,128] + bias + add )
// bf16 WMMA (v_wmma_f32_16x16x32_bf16), fp32 accumulate.
// Block: 256 threads = 8 waves; each wave owns 2 row-tiles (32 rows) -> 256 rows/block.
// W staged in LDS pre-swizzled into B-fragment order (v16bf per lane per tile).
// ---------------------------------------------------------------------------
__global__ __launch_bounds__(256)
void gemm128_bf16(const float* __restrict__ in, const float* __restrict__ W,
                  const float* __restrict__ bias, const float* __restrict__ add,
                  float* __restrict__ out, int M, int relu)
{
    __shared__ v16bf Bfrag[4][8][32];   // [kTile][colTile][lane] = 32 KB
    __shared__ float biasS[DHID];

    const int tid   = threadIdx.x;
    const int lane  = tid & 31;
    const int wave  = tid >> 5;
    const int group = lane >> 4;       // lane half (0: lanes 0-15, 1: lanes 16-31)
    const int laneM = lane & 15;

    // ---- stage W into LDS in fragment order (B: 32x16 K-major tile per WMMA) ----
    for (int s = tid; s < 4 * 8 * 32; s += 256) {
        const int kt  = s >> 8;
        const int ct  = (s >> 5) & 7;
        const int ln  = s & 31;
        const int grp = ln >> 4;
        const int n   = ct * 16 + (ln & 15);
        v16bf frag;
#pragma unroll
        for (int i = 0; i < 16; ++i) {
            const int k = kt * 32 + 8 * grp + (i < 8 ? i : i + 8);
            frag[i] = (__bf16)W[k * DHID + n];
        }
        Bfrag[kt][ct][ln] = frag;
    }
    if (tid < DHID) biasS[tid] = bias ? bias[tid] : 0.0f;
    __syncthreads();

    const long long rowBase = (long long)blockIdx.x * 256 + wave * 32;

    v8f acc[2][8];
#pragma unroll
    for (int rt = 0; rt < 2; ++rt)
#pragma unroll
        for (int ct = 0; ct < 8; ++ct) acc[rt][ct] = v8f{};

#pragma unroll
    for (int kt = 0; kt < 4; ++kt) {
        v16bf a[2];
#pragma unroll
        for (int rt = 0; rt < 2; ++rt) {
            long long r = rowBase + rt * 16 + laneM;
            if (r >= M) r = M - 1;                       // clamp (rows never stored)
            const float* rowp = in + r * DHID + kt * 32 + 8 * group;
#pragma unroll
            for (int i = 0; i < 8; ++i) a[rt][i]     = (__bf16)rowp[i];
#pragma unroll
            for (int i = 0; i < 8; ++i) a[rt][i + 8] = (__bf16)rowp[16 + i];
        }
#pragma unroll
        for (int ct = 0; ct < 8; ++ct) {
            const v16bf b = Bfrag[kt][ct][lane];
#pragma unroll
            for (int rt = 0; rt < 2; ++rt)
                acc[rt][ct] = __builtin_amdgcn_wmma_f32_16x16x32_bf16(
                    false, a[rt], false, b, (short)0, acc[rt][ct], false, false);
        }
    }

    // ---- epilogue: C/D layout M = i + 8*group, N = laneM ----
#pragma unroll
    for (int rt = 0; rt < 2; ++rt) {
#pragma unroll
        for (int i = 0; i < 8; ++i) {
            const long long r = rowBase + rt * 16 + 8 * group + i;
            if (r < M) {
                float* orow = out + r * DHID;
                const float* arow = add ? add + r * DHID : nullptr;
#pragma unroll
                for (int ct = 0; ct < 8; ++ct) {
                    const int c = ct * 16 + laneM;
                    float val = acc[rt][ct][i] + biasS[c];
                    if (arow) val += arow[c];
                    if (relu) val = fmaxf(val, 0.0f);
                    orow[c] = val;
                }
            }
        }
    }
}

// ---------------------------------------------------------------------------
// Attention helpers
// ---------------------------------------------------------------------------
__device__ inline void atomicMaxFloat(float* addr, float val)
{
    if (val >= 0.0f) atomicMax((int*)addr, __float_as_int(val));
    else             atomicMin((unsigned int*)addr, __float_as_uint(val));
}

__global__ void attn_init_kernel(float* __restrict__ smax, float* __restrict__ denom, int n)
{
    const int i = blockIdx.x * blockDim.x + threadIdx.x;
    if (i < n) { smax[i] = -__builtin_inff(); denom[i] = 0.0f; }
}

// warp per edge: score[e,h] = dot32(q[dst[e],h], k[e,h]) * rsqrt(32); segment max
__global__ void attn_score_kernel(const float* __restrict__ q, const float* __restrict__ k,
                                  const int* __restrict__ dst, float* __restrict__ scores,
                                  float* __restrict__ smax, int E)
{
    const int e = (blockIdx.x * blockDim.x + threadIdx.x) >> 5;
    if (e >= E) return;
    const int lane = threadIdx.x & 31;
    const int d = dst[e];
    const float4 qv = ((const float4*)(q + (long long)d * DHID))[lane];
    const float4 kv = ((const float4*)(k + (long long)e * DHID))[lane];
    float s = qv.x * kv.x + qv.y * kv.y + qv.z * kv.z + qv.w * kv.w;
    s += __shfl_xor(s, 1, 32);
    s += __shfl_xor(s, 2, 32);
    s += __shfl_xor(s, 4, 32);
    if ((lane & 7) == 0) {
        const int hh = lane >> 3;
        const float sc = s * 0.1767766952966369f;   // 1/sqrt(32)
        scores[(long long)e * 4 + hh] = sc;
        atomicMaxFloat(&smax[d * 4 + hh], sc);
    }
}

// warp per edge: p = exp(score - smax); denom += p; ctx += p * v
__global__ void attn_accum_kernel(const float* __restrict__ v, const int* __restrict__ dst,
                                  const float* __restrict__ scores, const float* __restrict__ smax,
                                  float* __restrict__ denom, float* __restrict__ ctx, int E)
{
    const int e = (blockIdx.x * blockDim.x + threadIdx.x) >> 5;
    if (e >= E) return;
    const int lane = threadIdx.x & 31;
    const int hh = lane >> 3;
    const int d = dst[e];
    const float p = __expf(scores[(long long)e * 4 + hh] - smax[d * 4 + hh]);
    if ((lane & 7) == 0) atomicAdd(&denom[d * 4 + hh], p);
    const float4 vv = ((const float4*)(v + (long long)e * DHID))[lane];
    float* cp = ctx + (long long)d * DHID + lane * 4;
    atomicAdd(cp + 0, p * vv.x);
    atomicAdd(cp + 1, p * vv.y);
    atomicAdd(cp + 2, p * vv.z);
    atomicAdd(cp + 3, p * vv.w);
}

__global__ void ctx_norm_kernel(float* __restrict__ ctx, const float* __restrict__ denom, int n)
{
    const int t = blockIdx.x * blockDim.x + threadIdx.x;
    if (t < n * DHID) {
        const int node = t >> 7;
        const int hh = (t & 127) >> 5;
        const float dn = denom[node * 4 + hh];
        ctx[t] = dn > 0.0f ? ctx[t] / dn : 0.0f;
    }
}

// m[e] = f_h[src[e]] - h[e^1]   (reverse pair = interleaved neighbour)
__global__ void edge_msg_kernel(const float* __restrict__ f_h, const float* __restrict__ h,
                                const int* __restrict__ src, float* __restrict__ m, int E)
{
    const long long t = (long long)blockIdx.x * blockDim.x + threadIdx.x;
    if (t < (long long)E * 32) {
        const long long e = t >> 5;
        const int c4 = (int)(t & 31);
        const int s = src[e];
        const float4 a = ((const float4*)(f_h + (long long)s * DHID))[c4];
        const float4 b = ((const float4*)(h + (e ^ 1) * DHID))[c4];
        ((float4*)(m + e * DHID))[c4] = make_float4(a.x - b.x, a.y - b.y, a.z - b.z, a.w - b.w);
    }
}

__global__ void mail_kernel(const float* __restrict__ h, const int* __restrict__ dst,
                            float* __restrict__ mail, int E)
{
    const long long t = (long long)blockIdx.x * blockDim.x + threadIdx.x;
    if (t < (long long)E * 32) {
        const long long e = t >> 5;
        const int c4 = (int)(t & 31);
        const int d = dst[e];
        const float4 hv = ((const float4*)(h + e * DHID))[c4];
        float* mp = mail + (long long)d * DHID + c4 * 4;
        atomicAdd(mp + 0, hv.x);
        atomicAdd(mp + 1, hv.y);
        atomicAdd(mp + 2, hv.z);
        atomicAdd(mp + 3, hv.w);
    }
}

// ---------------------------------------------------------------------------
extern "C" void kernel_launch(void* const* d_in, const int* in_sizes, int n_in,
                              void* d_out, int out_size, void* d_ws, size_t ws_size,
                              hipStream_t stream)
{
    const float* f      = (const float*)d_in[0];
    const float* x      = (const float*)d_in[1];
    const int*   src    = (const int*)d_in[2];
    const int*   dst    = (const int*)d_in[3];
    const float* Wq     = (const float*)d_in[4];
    const float* bq     = (const float*)d_in[5];
    const float* Wk     = (const float*)d_in[6];
    const float* bk     = (const float*)d_in[7];
    const float* Wv     = (const float*)d_in[8];
    const float* bv     = (const float*)d_in[9];
    const float* Wo     = (const float*)d_in[10];
    const float* bo     = (const float*)d_in[11];
    const float* W_mp   = (const float*)d_in[12];
    const float* b_mp   = (const float*)d_in[13];
    const float* W_last = (const float*)d_in[14];
    const float* b_last = (const float*)d_in[15];

    const int N = in_sizes[0] / DHID;
    const int E = in_sizes[2];

    // workspace layout (fp32)
    float* p = (float*)d_ws;
    float* f_h    = p; p += (long long)N * DHID;
    float* h      = p; p += (long long)E * DHID;
    float* kbuf   = p; p += (long long)E * DHID;   // reused as m
    float* vbuf   = p; p += (long long)E * DHID;
    float* qbuf   = p; p += (long long)N * DHID;
    float* ctx    = p; p += (long long)N * DHID;
    float* mail   = p; p += (long long)N * DHID;
    float* scores = p; p += (long long)E * 4;
    float* smax   = p; p += (long long)N * 4;
    float* denom  = p; p += (long long)N * 4;
    float* out    = (float*)d_out;

    hipMemcpyAsync(f_h, f, (size_t)N * DHID * sizeof(float), hipMemcpyDeviceToDevice, stream);
    hipMemcpyAsync(h,   x, (size_t)E * DHID * sizeof(float), hipMemcpyDeviceToDevice, stream);

    auto gemm = [&](const float* in, const float* W, const float* bias,
                    const float* add, float* o, int M, int relu) {
        const int blocks = (M + 255) / 256;
        gemm128_bf16<<<blocks, 256, 0, stream>>>(in, W, bias, add, o, M, relu);
    };

    const int warpBlocks = (int)(((long long)E * 32 + 255) / 256);

    for (int it = 0; it < 2; ++it) {
        // q/k/v projections (WMMA GEMMs)
        gemm(f_h, Wq, bq, nullptr, qbuf, N, 0);
        gemm(h,   Wk, bk, nullptr, kbuf, E, 0);
        gemm(h,   Wv, bv, nullptr, vbuf, E, 0);

        // softmax over incoming edges per node/head
        attn_init_kernel<<<(N * 4 + 255) / 256, 256, 0, stream>>>(smax, denom, N * 4);
        hipMemsetAsync(ctx, 0, (size_t)N * DHID * sizeof(float), stream);
        attn_score_kernel<<<(E + 7) / 8, 256, 0, stream>>>(qbuf, kbuf, dst, scores, smax, E);
        attn_accum_kernel<<<(E + 7) / 8, 256, 0, stream>>>(vbuf, dst, scores, smax, denom, ctx, E);
        ctx_norm_kernel<<<(N * DHID + 255) / 256, 256, 0, stream>>>(ctx, denom, N);

        // f_h = ctx @ Wo + bo + f_h   (in-place residual, element-wise safe)
        gemm(ctx, Wo, bo, f_h, f_h, N, 0);

        // m = f_h[src] - rev_h ; h = relu(x + m @ W_mp[it] + b_mp[it])
        edge_msg_kernel<<<warpBlocks, 256, 0, stream>>>(f_h, h, src, kbuf, E);
        gemm(kbuf, W_mp + (long long)it * DHID * DHID, b_mp + it * DHID, x, h, E, 1);
    }

    // mail = segment_sum(h, dst); out = [mail | f_h | f] @ W_last + b_last
    hipMemsetAsync(mail, 0, (size_t)N * DHID * sizeof(float), stream);
    mail_kernel<<<warpBlocks, 256, 0, stream>>>(h, dst, mail, E);

    gemm(mail, W_last,                  b_last, nullptr, out, N, 0);
    gemm(f_h,  W_last + 128 * DHID,     nullptr, out,    out, N, 0);
    gemm(f,    W_last + 256 * DHID,     nullptr, out,    out, N, 0);
}
